// Model_39676907885919
// MI455X (gfx1250) — matrix operand order repaired and checked by reading, not compiled, not run on hardware
//
#include <hip/hip_runtime.h>
#include <hip/hip_bf16.h>

typedef __attribute__((ext_vector_type(16))) _Float16 v16h;
typedef __attribute__((ext_vector_type(8)))  _Float16 v8h;
typedef __attribute__((ext_vector_type(2)))  __fp16   g2;   // type returned by cvt_pkrtz
typedef __attribute__((ext_vector_type(4)))  __fp16   g4;
typedef __attribute__((ext_vector_type(8)))  float    v8f;

#define BATCH 2
#define HEADS 16
#define SEQ   2048
#define HDIM  64
#define QTILE 128     // query rows per workgroup (8 waves x 16)
#define KT    64      // keys per inner iteration
#define NWAVE 8
#define NT    (SEQ / KT)

// butterfly xor-shuffle within each row of 16 lanes, pure VALU (v_permlane16_b32)
__device__ __forceinline__ float bfly16(float v, unsigned s0, unsigned s1) {
    unsigned u = __float_as_uint(v);
    u = __builtin_amdgcn_permlane16(u, u, s0, s1, false, false);
    return __uint_as_float(u);
}
// nibble-select tables for lane -> lane^m (per row of 16)
#define X1_LO 0x67452301u
#define X1_HI 0xEFCDAB89u
#define X2_LO 0x54761032u
#define X2_HI 0xDCFE98BAu
#define X4_LO 0x32107654u
#define X4_HI 0xBA98FEDCu
#define X8_LO 0xFEDCBA98u
#define X8_HI 0x76543210u

__device__ __forceinline__ float rowmax16(float v) {
    v = fmaxf(v, bfly16(v, X1_LO, X1_HI));
    v = fmaxf(v, bfly16(v, X2_LO, X2_HI));
    v = fmaxf(v, bfly16(v, X4_LO, X4_HI));
    v = fmaxf(v, bfly16(v, X8_LO, X8_HI));
    return v;
}
__device__ __forceinline__ float rowsum16(float v) {
    v += bfly16(v, X1_LO, X1_HI);
    v += bfly16(v, X2_LO, X2_HI);
    v += bfly16(v, X4_LO, X4_HI);
    v += bfly16(v, X8_LO, X8_HI);
    return v;
}

// D = softmax(scale_b * Q K^T) ; out = (D * key_mask) @ V   (flash-attention style)
// scale folded into Q at staging; key_mask folded into V at staging.
__global__ __launch_bounds__(256)
void Model_attn_fa_kernel(const float* __restrict__ Q,
                          const float* __restrict__ K,
                          const float* __restrict__ V,
                          const float* __restrict__ keymask,   // [B, S]
                          const float* __restrict__ scalef,    // [B]
                          float* __restrict__ Out)
{
    __shared__ _Float16 sQ[QTILE * HDIM];         // 16 KB  [qrow][d]  (pre-scaled)
    __shared__ _Float16 sK[2][KT * HDIM];         // 16 KB  [key][d]   double-buffered
    __shared__ _Float16 sVt[2][HDIM * KT];        // 16 KB  [d][key]   (masked, transposed)
    __shared__ _Float16 sP[NWAVE][16 * KT];       // 16 KB  per-wave P bounce [m][k]

    const int nqblk = SEQ / QTILE;             // 16
    const int bh    = blockIdx.x / nqblk;      // 0..B*H-1
    const int qblk  = blockIdx.x % nqblk;
    const int b     = bh / HEADS;
    const int tid   = threadIdx.x;
    const int w     = tid >> 5;                // wave id 0..7
    const int lane  = tid & 31;
    const int l16   = lane & 15;
    const int hi    = lane >> 4;               // half-wave: 0 or 1

    // fold log2(e) into the per-batch scale: softmax runs in exp2 domain
    const float  scale2 = scalef[b] * 1.4426950408889634f;
    const size_t base   = (size_t)bh * SEQ * HDIM;
    const float* Qg = Q + base + (size_t)qblk * QTILE * HDIM;
    const float* Kg = K + base;
    const float* Vg = V + base;
    const float* km = keymask + (size_t)b * SEQ;

    // ---- stage Q tile: 128x64 f32 -> (scale2*Q) f16 LDS ----
    {
        const float4* q4 = (const float4*)Qg;   // 2048 float4 chunks
        #pragma unroll
        for (int i = 0; i < 8; ++i) {
            int c = tid + i * 256;
            float4 v = q4[c];
            g2 p01 = __builtin_amdgcn_cvt_pkrtz(v.x * scale2, v.y * scale2);
            g2 p23 = __builtin_amdgcn_cvt_pkrtz(v.z * scale2, v.w * scale2);
            *(g4*)&sQ[c * 4] = __builtin_shufflevector(p01, p23, 0, 1, 2, 3);
        }
    }

    // staging registers for the in-flight K/V tile
    float4 kr[4], vr[4];
    float  kmr[4];

    // issue + stage tile 0 into buffer 0
    {
        const float4* k4 = (const float4*)Kg;
        const float4* v4 = (const float4*)Vg;
        #pragma unroll
        for (int i = 0; i < 4; ++i) {
            int c = tid + i * 256;
            kr[i]  = k4[c];
            vr[i]  = v4[c];
            kmr[i] = km[c >> 4];
        }
        #pragma unroll
        for (int i = 0; i < 4; ++i) {
            int c = tid + i * 256;
            g2 k01 = __builtin_amdgcn_cvt_pkrtz(kr[i].x, kr[i].y);
            g2 k23 = __builtin_amdgcn_cvt_pkrtz(kr[i].z, kr[i].w);
            *(g4*)&sK[0][c * 4] = __builtin_shufflevector(k01, k23, 0, 1, 2, 3);
            int key = c >> 4;                  // 0..63
            int dp  = (c & 15) * 4;
            float mk = kmr[i];
            sVt[0][(dp + 0) * KT + key] = (_Float16)(vr[i].x * mk);
            sVt[0][(dp + 1) * KT + key] = (_Float16)(vr[i].y * mk);
            sVt[0][(dp + 2) * KT + key] = (_Float16)(vr[i].z * mk);
            sVt[0][(dp + 3) * KT + key] = (_Float16)(vr[i].w * mk);
        }
    }
    __syncthreads();

    // ---- Q A-fragments (16x32 f16 each), loop-invariant.
    // A layout: lanes0-15 (M=l16): K = {0..15}; lanes16-31: K = {16..31}
    v16h aq0, aq1;
    {
        const _Float16* qrow = &sQ[(w * 16 + l16) * HDIM];
        v8h a = *(const v8h*)(qrow + hi * 16);
        v8h bfr = *(const v8h*)(qrow + 8 + hi * 16);
        aq0 = __builtin_shufflevector(a, bfr, 0,1,2,3,4,5,6,7,8,9,10,11,12,13,14,15);
        v8h c = *(const v8h*)(qrow + 32 + hi * 16);
        v8h d = *(const v8h*)(qrow + 40 + hi * 16);
        aq1 = __builtin_shufflevector(c, d, 0,1,2,3,4,5,6,7,8,9,10,11,12,13,14,15);
    }

    // per-row online-softmax state (log2 domain); element r -> row M = r + 8*hi
    float mrow[8], lrow[8];
    #pragma unroll
    for (int r = 0; r < 8; ++r) { mrow[r] = -1e30f; lrow[r] = 0.0f; }
    v8f o0 = {}, o1 = {}, o2 = {}, o3 = {};    // O accumulators, d ranges 0/16/32/48

    #pragma unroll 1
    for (int kt = 0; kt < NT; ++kt) {
        const int  cur  = kt & 1;
        const bool more = (kt + 1) < NT;

        // ---- issue next tile's global loads (latency hidden under WMMAs) ----
        if (more) {
            const float4* k4 = (const float4*)Kg + (size_t)(kt + 1) * (KT * HDIM / 4);
            const float4* v4 = (const float4*)Vg + (size_t)(kt + 1) * (KT * HDIM / 4);
            #pragma unroll
            for (int i = 0; i < 4; ++i) {
                int c = tid + i * 256;
                kr[i]  = k4[c];
                vr[i]  = v4[c];
                kmr[i] = km[(kt + 1) * KT + (c >> 4)];
            }
            if (kt + 2 < NT) {                 // global_prefetch_b8 for tile kt+2
                __builtin_prefetch((const void*)(k4 + KT * HDIM / 4 + tid * 4), 0, 0);
                __builtin_prefetch((const void*)(v4 + KT * HDIM / 4 + tid * 4), 0, 0);
            }
        }

        // ---- scores: four 16x16 C tiles; Q was pre-scaled -> log2-domain directly ----
        v8f c0 = {}, c1 = {}, c2 = {}, c3 = {};
        {
            const _Float16* kb = &sK[cur][l16 * HDIM + hi * 16];
            v16h bkA, bkB;
            bkA = *(const v16h*)(kb);
            bkB = *(const v16h*)(kb + 32);
            c0 = __builtin_amdgcn_wmma_f32_16x16x32_f16(false, aq0, false, bkA, (short)0, c0, false, false);
            c0 = __builtin_amdgcn_wmma_f32_16x16x32_f16(false, aq1, false, bkB, (short)0, c0, false, false);
            bkA = *(const v16h*)(kb + 16 * HDIM);
            bkB = *(const v16h*)(kb + 16 * HDIM + 32);
            c1 = __builtin_amdgcn_wmma_f32_16x16x32_f16(false, aq0, false, bkA, (short)0, c1, false, false);
            c1 = __builtin_amdgcn_wmma_f32_16x16x32_f16(false, aq1, false, bkB, (short)0, c1, false, false);
            bkA = *(const v16h*)(kb + 32 * HDIM);
            bkB = *(const v16h*)(kb + 32 * HDIM + 32);
            c2 = __builtin_amdgcn_wmma_f32_16x16x32_f16(false, aq0, false, bkA, (short)0, c2, false, false);
            c2 = __builtin_amdgcn_wmma_f32_16x16x32_f16(false, aq1, false, bkB, (short)0, c2, false, false);
            bkA = *(const v16h*)(kb + 48 * HDIM);
            bkB = *(const v16h*)(kb + 48 * HDIM + 32);
            c3 = __builtin_amdgcn_wmma_f32_16x16x32_f16(false, aq0, false, bkA, (short)0, c3, false, false);
            c3 = __builtin_amdgcn_wmma_f32_16x16x32_f16(false, aq1, false, bkB, (short)0, c3, false, false);
        }

        float alpha[8];
        #pragma unroll
        for (int r = 0; r < 8; ++r) {
            float s0 = c0[r], s1 = c1[r], s2 = c2[r], s3 = c3[r];
            float mx   = rowmax16(fmaxf(fmaxf(s0, s1), fmaxf(s2, s3)));
            float mnew = fmaxf(mrow[r], mx);
            float a    = __builtin_amdgcn_exp2f(mrow[r] - mnew);
            float p0   = __builtin_amdgcn_exp2f(s0 - mnew);
            float p1   = __builtin_amdgcn_exp2f(s1 - mnew);
            float p2   = __builtin_amdgcn_exp2f(s2 - mnew);
            float p3   = __builtin_amdgcn_exp2f(s3 - mnew);
            float rs   = rowsum16((p0 + p1) + (p2 + p3));  // denominator: UNMASKED
            lrow[r]  = lrow[r] * a + rs;
            mrow[r]  = mnew;
            alpha[r] = a;
            // raw P -> per-wave LDS bounce (mask folded into V)
            int m = r + hi * 8;
            sP[w][m * KT + l16]      = (_Float16)p0;
            sP[w][m * KT + 16 + l16] = (_Float16)p1;
            sP[w][m * KT + 32 + l16] = (_Float16)p2;
            sP[w][m * KT + 48 + l16] = (_Float16)p3;
        }
        asm volatile("s_wait_dscnt 0" ::: "memory");

        // P A-fragments (16x64 -> two 16x32): same A layout as Q
        v16h ap0, ap1;
        {
            const _Float16* prow = &sP[w][l16 * KT];
            v8h a = *(const v8h*)(prow + hi * 16);
            v8h bfr = *(const v8h*)(prow + 8 + hi * 16);
            ap0 = __builtin_shufflevector(a, bfr, 0,1,2,3,4,5,6,7,8,9,10,11,12,13,14,15);
            v8h c = *(const v8h*)(prow + 32 + hi * 16);
            v8h d = *(const v8h*)(prow + 40 + hi * 16);
            ap1 = __builtin_shufflevector(c, d, 0,1,2,3,4,5,6,7,8,9,10,11,12,13,14,15);
        }

        // rescale O accumulators by alpha per row
        #pragma unroll
        for (int r = 0; r < 8; ++r) {
            o0[r] *= alpha[r]; o1[r] *= alpha[r];
            o2[r] *= alpha[r]; o3[r] *= alpha[r];
        }

        // ---- PV: O(16x64) += P(16x64) @ (mask*V)(64x64) ----
        {
            const _Float16* vb = &sVt[cur][0];
            v16h bvA, bvB;
            bvA = *(const v16h*)&vb[( 0 + l16) * KT + hi * 16];
            bvB = *(const v16h*)&vb[( 0 + l16) * KT + 32 + hi * 16];
            o0 = __builtin_amdgcn_wmma_f32_16x16x32_f16(false, ap0, false, bvA, (short)0, o0, false, false);
            o0 = __builtin_amdgcn_wmma_f32_16x16x32_f16(false, ap1, false, bvB, (short)0, o0, false, false);
            bvA = *(const v16h*)&vb[(16 + l16) * KT + hi * 16];
            bvB = *(const v16h*)&vb[(16 + l16) * KT + 32 + hi * 16];
            o1 = __builtin_amdgcn_wmma_f32_16x16x32_f16(false, ap0, false, bvA, (short)0, o1, false, false);
            o1 = __builtin_amdgcn_wmma_f32_16x16x32_f16(false, ap1, false, bvB, (short)0, o1, false, false);
            bvA = *(const v16h*)&vb[(32 + l16) * KT + hi * 16];
            bvB = *(const v16h*)&vb[(32 + l16) * KT + 32 + hi * 16];
            o2 = __builtin_amdgcn_wmma_f32_16x16x32_f16(false, ap0, false, bvA, (short)0, o2, false, false);
            o2 = __builtin_amdgcn_wmma_f32_16x16x32_f16(false, ap1, false, bvB, (short)0, o2, false, false);
            bvA = *(const v16h*)&vb[(48 + l16) * KT + hi * 16];
            bvB = *(const v16h*)&vb[(48 + l16) * KT + 32 + hi * 16];
            o3 = __builtin_amdgcn_wmma_f32_16x16x32_f16(false, ap0, false, bvA, (short)0, o3, false, false);
            o3 = __builtin_amdgcn_wmma_f32_16x16x32_f16(false, ap1, false, bvB, (short)0, o3, false, false);
        }

        // ---- stage next tile into the other buffer (loads have had time to land) ----
        if (more) {
            const int nxt = cur ^ 1;
            #pragma unroll
            for (int i = 0; i < 4; ++i) {
                int c = tid + i * 256;
                g2 k01 = __builtin_amdgcn_cvt_pkrtz(kr[i].x, kr[i].y);
                g2 k23 = __builtin_amdgcn_cvt_pkrtz(kr[i].z, kr[i].w);
                *(g4*)&sK[nxt][c * 4] = __builtin_shufflevector(k01, k23, 0, 1, 2, 3);
                int key = c >> 4;
                int dp  = (c & 15) * 4;
                float mk = kmr[i];
                sVt[nxt][(dp + 0) * KT + key] = (_Float16)(vr[i].x * mk);
                sVt[nxt][(dp + 1) * KT + key] = (_Float16)(vr[i].y * mk);
                sVt[nxt][(dp + 2) * KT + key] = (_Float16)(vr[i].z * mk);
                sVt[nxt][(dp + 3) * KT + key] = (_Float16)(vr[i].w * mk);
            }
        }
        __syncthreads();    // next buffer staged everywhere; cur free for reuse
    }

    // ---- epilogue: out = O / l ----
    float* outp = Out + base + (size_t)(qblk * QTILE + w * 16) * HDIM;
    #pragma unroll
    for (int r = 0; r < 8; ++r) {
        int   m   = r + hi * 8;
        float inv = 1.0f / lrow[r];
        outp[m * HDIM +      l16] = o0[r] * inv;
        outp[m * HDIM + 16 + l16] = o1[r] * inv;
        outp[m * HDIM + 32 + l16] = o2[r] * inv;
        outp[m * HDIM + 48 + l16] = o3[r] * inv;
    }
}

extern "C" void kernel_launch(void* const* d_in, const int* in_sizes, int n_in,
                              void* d_out, int out_size, void* d_ws, size_t ws_size,
                              hipStream_t stream) {
    (void)in_sizes; (void)n_in; (void)out_size; (void)d_ws; (void)ws_size;
    const float* Q     = (const float*)d_in[0];   // [B,H,S,D]
    const float* K     = (const float*)d_in[1];   // [B,H,S,D]
    const float* V     = (const float*)d_in[2];   // [B,H,S,D]
    /* d_in[3] = query_mask: unused by reference */
    const float* kmask = (const float*)d_in[4];   // [B,1,1,S]
    const float* scal  = (const float*)d_in[5];   // [B,1,1,1]
    /* d_in[6] = dropout (0): identity */
    float* Out = (float*)d_out;

    const int nblocks = BATCH * HEADS * (SEQ / QTILE);  // 512
    Model_attn_fa_kernel<<<nblocks, 256, 0, stream>>>(Q, K, V, kmask, scal, Out);
}
